// GQA_17257178595392
// MI455X (gfx1250) — matrix operand order repaired and checked by reading, not compiled, not run on hardware
//
#include <hip/hip_runtime.h>

// ---------------------------------------------------------------------------
// GQA layer for MI455X (gfx1250, wave32, WMMA).
// All matmuls run through v_wmma_f32_16x16x32_bf16 (fp32 inputs downcast to
// bf16, fp32 accumulate). Compute-bound workload: ~77 GFLOP vs ~100MB bytes.
// GEMM wave tile is 32x64 (8 accumulators) -> 1.5 b128 loads per WMMA.
// ---------------------------------------------------------------------------

typedef __attribute__((ext_vector_type(16))) __bf16 v16bf;
typedef __attribute__((ext_vector_type(8)))  __bf16 v8bf;
typedef __attribute__((ext_vector_type(8)))  float  v8f;

#define D_MODEL 2048
#define SEQ     2048
#define HQN     32
#define HKN     8
#define DKN     64
#define KV_DIM  512   // HKN*DKN

__device__ __forceinline__ __bf16 f2bf(float f) {
  // round-to-nearest-even fp32 -> bf16 (bit manipulation; no cvt dependency)
  unsigned u = __builtin_bit_cast(unsigned, f);
  u += 0x7FFFu + ((u >> 16) & 1u);
  unsigned short s = (unsigned short)(u >> 16);
  return __builtin_bit_cast(__bf16, s);
}

union AFrag { v16bf v; v8bf h[2]; };

// ---------------------------------------------------------------------------
// fp32 -> bf16 elementwise (weights)
// ---------------------------------------------------------------------------
__global__ void f32_to_bf16_kernel(const float* __restrict__ src,
                                   __bf16* __restrict__ dst, long long n) {
  long long i = (long long)blockIdx.x * blockDim.x + threadIdx.x;
  long long stride = (long long)gridDim.x * blockDim.x;
  for (; i < n; i += stride) dst[i] = f2bf(src[i]);
}

// ---------------------------------------------------------------------------
// embedding gather + bf16 convert: e[s][d] = emb[X[s]][d]
// ---------------------------------------------------------------------------
__global__ void embed_gather_bf16(const int* __restrict__ X,
                                  const float* __restrict__ emb,
                                  __bf16* __restrict__ e) {
  int s = blockIdx.x;
  long long row = (long long)X[s];
  const float* src = emb + row * D_MODEL;
  __bf16* dst = e + (long long)s * D_MODEL;
  for (int d = threadIdx.x; d < D_MODEL; d += blockDim.x) dst[d] = f2bf(src[d]);
}

// ---------------------------------------------------------------------------
// WMMA GEMM:  out[M,N] = A[M,K] @ W[N,K]^T
//   A row-major bf16, W row-major bf16 ("torch Linear" weight layout, so each
//   lane's B-fragment is a contiguous 16-element run of one W row).
//   Wave tile: 32(M) x 64(N), acc[2][4]; each B fragment feeds 2 WMMAs.
//   Block = 4 waves in a 2(M) x 2(N) arrangement -> 64 x 128 block tile.
//   Output modes: bf16 row-major | bf16 transposed (for V^T) | fp32 + bias.
// ---------------------------------------------------------------------------
__global__ __launch_bounds__(128)
void gemm_bf16_wmma(const __bf16* __restrict__ A, const __bf16* __restrict__ W,
                    int M, int N, int K,
                    __bf16* outB, float* outF, const float* __restrict__ bias,
                    int transposeOut, int ldOut) {
  const int lane  = threadIdx.x & 31;
  const int wave  = threadIdx.x >> 5;
  const int waveM = wave & 1;
  const int waveN = wave >> 1;
  const int half  = lane >> 4;    // K-half selector (this code's contiguous k = 16*half + s order)
  const int l16   = lane & 15;

  const int m0 = blockIdx.y * 64 + waveM * 32;   // wave owns rows m0..m0+31
  const int n0 = blockIdx.x * 128 + waveN * 64;  // wave owns cols n0..n0+63

  // A: lane holds row; per 32-chunk it reads k in {b..b+15} (contiguous 16-run)
  const __bf16* Arow0 = A + (long long)(m0 + l16) * K + half * 16;
  const __bf16* Arow1 = A + (long long)(m0 + 16 + l16) * K + half * 16;
  // B: lane holds output column; contiguous 16-elem K-run at half*16
  const __bf16* Wl[4];
#pragma unroll
  for (int j = 0; j < 4; ++j)
    Wl[j] = W + (long long)(n0 + j * 16 + l16) * K + half * 16;

  v8f acc[2][4] = {};

  for (int k0 = 0; k0 < K; k0 += 32) {
    AFrag a0, a1;
    a0.h[0] = *(const v8bf*)(Arow0 + k0);
    a0.h[1] = *(const v8bf*)(Arow0 + k0 + 8);
    a1.h[0] = *(const v8bf*)(Arow1 + k0);
    a1.h[1] = *(const v8bf*)(Arow1 + k0 + 8);
#pragma unroll
    for (int j = 0; j < 4; ++j) {
      v16bf b = *(const v16bf*)(Wl[j] + k0);
      acc[0][j] = __builtin_amdgcn_wmma_f32_16x16x32_bf16(
          false, a0.v, false, b, (short)0, acc[0][j], false, false);
      acc[1][j] = __builtin_amdgcn_wmma_f32_16x16x32_bf16(
          false, a1.v, false, b, (short)0, acc[1][j], false, false);
    }
  }

  // C layout: lane -> (m = r + 8*half, n = l16) per VGPR r
#pragma unroll
  for (int i = 0; i < 2; ++i) {
#pragma unroll
    for (int j = 0; j < 4; ++j) {
#pragma unroll
      for (int r = 0; r < 8; ++r) {
        int row = m0 + i * 16 + r + 8 * half;
        int col = n0 + j * 16 + l16;
        float v = acc[i][j][r];
        if (outF) {
          outF[(long long)row * ldOut + col] = v + (bias ? bias[col] : 0.0f);
        } else if (transposeOut) {
          outB[(long long)col * ldOut + row] = f2bf(v);   // V^T store
        } else {
          outB[(long long)row * ldOut + col] = f2bf(v);
        }
      }
    }
  }
}

// ---------------------------------------------------------------------------
// Flash-style GQA attention, one wave = one q-head x 16 query rows.
//   Q  : [S, 2048] bf16 row-major (head hq occupies cols hq*64..hq*64+63)
//   Kb : [S, 512]  bf16 row-major (kv head k occupies cols k*64..)
//   Vt : [512, S]  bf16 (transposed V -> contiguous B-fragment loads)
//   Ctx: [S, 2048] bf16 row-major
// Online softmax over 64 chunks of 32 keys; P transposed C->A layout via LDS.
// ---------------------------------------------------------------------------
__global__ __launch_bounds__(128)
void attn_fa_wmma(const __bf16* __restrict__ Q, const __bf16* __restrict__ Kb,
                  const __bf16* __restrict__ Vt, __bf16* __restrict__ Ctx) {
  __shared__ alignas(16) __bf16 lds[4][16 * 48];  // per-wave P tile, stride 48

  const int lane = threadIdx.x & 31;
  const int wave = threadIdx.x >> 5;
  const int half = lane >> 4;
  const int l16  = lane & 15;

  const int hq  = blockIdx.x;          // 0..31 query head
  const int kvh = hq >> 2;             // grouped kv head (G=4)
  const int sq0 = (blockIdx.y * 4 + wave) * 16;

  __bf16* plds = lds[wave];

  // Q A-fragments for both 32-wide k-steps over d=0..63 (loop invariant)
  AFrag qa[2];
  {
    const __bf16* Qrow = Q + (long long)(sq0 + l16) * D_MODEL + hq * DKN + half * 16;
    qa[0].h[0] = *(const v8bf*)(Qrow + 0);
    qa[0].h[1] = *(const v8bf*)(Qrow + 8);
    qa[1].h[0] = *(const v8bf*)(Qrow + 32);
    qa[1].h[1] = *(const v8bf*)(Qrow + 40);
  }

  // K fragment base: lane's key column is t0 + nh*16 + l16 (rows of Kb)
  const __bf16* Kc = Kb + (long long)l16 * KV_DIM + kvh * DKN + half * 16;
  // V^T fragment row pointers (4 dv tiles of 16)
  const __bf16* Vrow[4];
#pragma unroll
  for (int j = 0; j < 4; ++j)
    Vrow[j] = Vt + (long long)(kvh * DKN + j * 16 + l16) * SEQ + half * 16;

  float m_run[8], l_run[8];
#pragma unroll
  for (int r = 0; r < 8; ++r) { m_run[r] = -__builtin_inff(); l_run[r] = 0.0f; }
  v8f acc[4] = {};

  const float sm = 0.125f;  // 1/sqrt(64)

  for (int t0 = 0; t0 < SEQ; t0 += 32) {
    // ---- S = Q K^T for 16 rows x 32 keys (two 16-col halves) ----
    v8f sfrag[2] = {};
#pragma unroll
    for (int dk = 0; dk < 2; ++dk) {
#pragma unroll
      for (int nh = 0; nh < 2; ++nh) {
        v16bf kb = *(const v16bf*)(Kc + (long long)(t0 + nh * 16) * KV_DIM + dk * 32);
        sfrag[nh] = __builtin_amdgcn_wmma_f32_16x16x32_bf16(
            false, qa[dk].v, false, kb, (short)0, sfrag[nh], false, false);
      }
    }

    // ---- online softmax (rows m = r + 8*half; N spread over 16 lanes) ----
    float corr[8];
#pragma unroll
    for (int r = 0; r < 8; ++r) {
      float s0 = sfrag[0][r] * sm;
      float s1 = sfrag[1][r] * sm;
      float mx = fmaxf(s0, s1);
      mx = fmaxf(mx, __shfl_xor(mx, 1));
      mx = fmaxf(mx, __shfl_xor(mx, 2));
      mx = fmaxf(mx, __shfl_xor(mx, 4));
      mx = fmaxf(mx, __shfl_xor(mx, 8));
      float mn = fmaxf(m_run[r], mx);
      float p0 = __expf(s0 - mn);
      float p1 = __expf(s1 - mn);
      float rs = p0 + p1;
      rs += __shfl_xor(rs, 1);
      rs += __shfl_xor(rs, 2);
      rs += __shfl_xor(rs, 4);
      rs += __shfl_xor(rs, 8);
      float c = __expf(m_run[r] - mn);
      l_run[r] = l_run[r] * c + rs;
      m_run[r] = mn;
      corr[r]  = c;
      // stage P (C layout) into LDS for the A-layout reload
      plds[(r + 8 * half) * 48 + l16]      = f2bf(p0);
      plds[(r + 8 * half) * 48 + 16 + l16] = f2bf(p1);
    }

    // reload P in A-fragment layout (wave-synchronous; DScnt ordered)
    AFrag pa;
    pa.h[0] = *(const v8bf*)(&plds[l16 * 48 + half * 16]);
    pa.h[1] = *(const v8bf*)(&plds[l16 * 48 + half * 16 + 8]);

    // ---- ctx = ctx*corr + P V ----
#pragma unroll
    for (int j = 0; j < 4; ++j) {
#pragma unroll
      for (int r = 0; r < 8; ++r) acc[j][r] *= corr[r];
      v16bf bv = *(const v16bf*)(Vrow[j] + t0);
      acc[j] = __builtin_amdgcn_wmma_f32_16x16x32_bf16(
          false, pa.v, false, bv, (short)0, acc[j], false, false);
    }
  }

  // ---- finalize: divide by l, store ctx bf16 ----
#pragma unroll
  for (int j = 0; j < 4; ++j) {
#pragma unroll
    for (int r = 0; r < 8; ++r) {
      float o = acc[j][r] / l_run[r];
      int row = sq0 + r + 8 * half;
      int col = hq * DKN + j * 16 + l16;
      Ctx[(long long)row * D_MODEL + col] = f2bf(o);
    }
  }
}

// ---------------------------------------------------------------------------
// Host-side launch
// ---------------------------------------------------------------------------
extern "C" void kernel_launch(void* const* d_in, const int* in_sizes, int n_in,
                              void* d_out, int out_size, void* d_ws, size_t ws_size,
                              hipStream_t stream) {
  const int*   X    = (const int*)d_in[0];
  const float* emb  = (const float*)d_in[1];
  const float* Wk   = (const float*)d_in[2];
  const float* Wv   = (const float*)d_in[3];
  const float* Wq   = (const float*)d_in[4];
  const float* Wo_w = (const float*)d_in[5];
  const float* Wo_b = (const float*)d_in[6];
  float* out = (float*)d_out;

  // workspace carve-up (bf16 buffers), all 256B-aligned by construction
  char* ws = (char*)d_ws;
  size_t off = 0;
  auto carve = [&](size_t bytes) { char* p = ws + off; off += (bytes + 255) & ~(size_t)255; return p; };
  __bf16* e_b   = (__bf16*)carve((size_t)SEQ * D_MODEL * 2);     // 8 MiB
  __bf16* q_b   = (__bf16*)carve((size_t)SEQ * D_MODEL * 2);     // 8 MiB
  __bf16* k_b   = (__bf16*)carve((size_t)SEQ * KV_DIM * 2);      // 2 MiB
  __bf16* vT_b  = (__bf16*)carve((size_t)KV_DIM * SEQ * 2);      // 2 MiB
  __bf16* ctx_b = (__bf16*)carve((size_t)SEQ * D_MODEL * 2);     // 8 MiB
  __bf16* wq_b  = (__bf16*)carve((size_t)D_MODEL * D_MODEL * 2); // 8 MiB
  __bf16* wk_b  = (__bf16*)carve((size_t)KV_DIM * D_MODEL * 2);  // 2 MiB
  __bf16* wv_b  = (__bf16*)carve((size_t)KV_DIM * D_MODEL * 2);  // 2 MiB
  __bf16* wo_b  = (__bf16*)carve((size_t)D_MODEL * D_MODEL * 2); // 8 MiB
  (void)ws_size; (void)in_sizes; (void)n_in; (void)out_size;

  // 1) weights -> bf16
  f32_to_bf16_kernel<<<1024, 256, 0, stream>>>(Wq,   wq_b, (long long)D_MODEL * D_MODEL);
  f32_to_bf16_kernel<<<1024, 256, 0, stream>>>(Wk,   wk_b, (long long)KV_DIM * D_MODEL);
  f32_to_bf16_kernel<<<1024, 256, 0, stream>>>(Wv,   wv_b, (long long)KV_DIM * D_MODEL);
  f32_to_bf16_kernel<<<1024, 256, 0, stream>>>(Wo_w, wo_b, (long long)D_MODEL * D_MODEL);

  // 2) embedding gather
  embed_gather_bf16<<<SEQ, 256, 0, stream>>>(X, emb, e_b);

  // 3) projections (WMMA GEMMs); block tile 64(M) x 128(N)
  dim3 blk(128);
  gemm_bf16_wmma<<<dim3(D_MODEL / 128, SEQ / 64), blk, 0, stream>>>(
      e_b, wq_b, SEQ, D_MODEL, D_MODEL, q_b, nullptr, nullptr, 0, D_MODEL);
  gemm_bf16_wmma<<<dim3(KV_DIM / 128, SEQ / 64), blk, 0, stream>>>(
      e_b, wk_b, SEQ, KV_DIM, D_MODEL, k_b, nullptr, nullptr, 0, KV_DIM);
  gemm_bf16_wmma<<<dim3(KV_DIM / 128, SEQ / 64), blk, 0, stream>>>(
      e_b, wv_b, SEQ, KV_DIM, D_MODEL, vT_b, nullptr, nullptr, 1, SEQ);  // V^T

  // 4) flash GQA attention (32 heads x 32 blocks of 4 query tiles)
  attn_fa_wmma<<<dim3(HQN, SEQ / 64), blk, 0, stream>>>(q_b, k_b, vT_b, ctx_b);

  // 5) output projection + bias -> fp32 d_out
  gemm_bf16_wmma<<<dim3(D_MODEL / 128, SEQ / 64), blk, 0, stream>>>(
      ctx_b, wo_b, SEQ, D_MODEL, D_MODEL, nullptr, out, Wo_b, 0, D_MODEL);
}
